// AttnBlock_25907242729939
// MI455X (gfx1250) — matrix operand order repaired and checked by reading, not compiled, not run on hardware
//
#include <hip/hip_runtime.h>
#include <hip/hip_bf16.h>

// CDNA5 / gfx1250 attention block. wave32, WMMA f32_16x16x32_f16, async-to-LDS staging.
typedef _Float16 half_t;
typedef __attribute__((ext_vector_type(16))) _Float16 v16h;
typedef __attribute__((ext_vector_type(8)))  _Float16 v8h;
typedef __attribute__((ext_vector_type(8)))  float    v8f;

#define B_   8
#define C_   256
#define HW   4096
#define G_   32
#define CPG  8
#define SCALE 0.0625f   // 256^-0.5

#define WMMA16(a, b, c) __builtin_amdgcn_wmma_f32_16x16x32_f16( \
    false, (a), false, (b), (short)0, (c), false, false)

// ---- CDNA5 async copy: one 16B chunk per lane, global -> LDS, tracked by ASYNCcnt ----
__device__ __forceinline__ void async_cp16(unsigned int lds_byte_addr, const void* gaddr) {
    asm volatile("global_load_async_to_lds_b128 %0, %1, off"
                 :: "v"(lds_byte_addr), "v"(gaddr)
                 : "memory");
}
__device__ __forceinline__ void wait_async0() {
#if __has_builtin(__builtin_amdgcn_s_wait_asynccnt)
    __builtin_amdgcn_s_wait_asynccnt(0);
#else
    asm volatile("s_wait_asynccnt 0x0" ::: "memory");
#endif
}
__device__ __forceinline__ unsigned int lds_addr32(const void* p) {
    // generic pointer to LDS: low 32 bits are the LDS byte offset (ISA 10.2)
    return (unsigned int)(unsigned long long)p;
}

// ---- fragment loaders (ISA 7.12.2 layouts) ----
// A 16x32 f16: lane l(0-15)/hi: row M=m0+l, K = {k0+hi*8..+7} and {k0+16+hi*8..+7}
__device__ __forceinline__ v16h frag_a(const half_t* base, int pitch, int m0, int k0, int lane) {
    const int l = lane & 15, hi = lane >> 4;
    const half_t* p = base + (size_t)(m0 + l) * pitch + k0 + hi * 8;
    v8h x0 = *(const v8h*)p;
    v8h x1 = *(const v8h*)(p + 16);
    v16h r;
#pragma unroll
    for (int i = 0; i < 8; ++i) { r[i] = x0[i]; r[i + 8] = x1[i]; }
    return r;
}
// B 32x16 f16 from [N][K] storage: lane l/hi: col N=n0+l, K = {k0+hi*8..+7} and {k0+16+hi*8..+7}
__device__ __forceinline__ v16h frag_b(const half_t* base, int pitch, int n0, int k0, int lane) {
    const int l = lane & 15, hi = lane >> 4;
    const half_t* p = base + (size_t)(n0 + l) * pitch + k0 + hi * 8;
    v8h x0 = *(const v8h*)p;
    v8h x1 = *(const v8h*)(p + 16);
    v16h r;
#pragma unroll
    for (int i = 0; i < 8; ++i) { r[i] = x0[i]; r[i + 8] = x1[i]; }
    return r;
}

// ---- 1) GroupNorm stats ----
__global__ void gn_stats_kernel(const float* __restrict__ x, float* __restrict__ stat) {
    __shared__ float s1[256], s2[256];
    const int bg = blockIdx.x;
    const float4* p = (const float4*)(x + (size_t)bg * (CPG * HW));
    float sum = 0.f, sq = 0.f;
    for (int i = threadIdx.x; i < (CPG * HW) / 4; i += 256) {
        float4 v = p[i];
        sum += v.x + v.y + v.z + v.w;
        sq  += v.x * v.x + v.y * v.y + v.z * v.z + v.w * v.w;
    }
    s1[threadIdx.x] = sum; s2[threadIdx.x] = sq;
    __syncthreads();
    for (int st = 128; st > 0; st >>= 1) {
        if (threadIdx.x < st) { s1[threadIdx.x] += s1[threadIdx.x + st]; s2[threadIdx.x] += s2[threadIdx.x + st]; }
        __syncthreads();
    }
    if (threadIdx.x == 0) {
        float inv_n = 1.0f / (float)(CPG * HW);
        float mean = s1[0] * inv_n;
        float var  = s2[0] * inv_n - mean * mean;
        stat[bg * 2 + 0] = mean;
        stat[bg * 2 + 1] = rsqrtf(var + 1e-5f);
    }
}

// ---- 2) GN apply -> hT[b][n][c] f16 ----
__global__ void gn_apply_kernel(const float* __restrict__ x, const float* __restrict__ gnw,
                                const float* __restrict__ gnb, const float* __restrict__ stat,
                                half_t* __restrict__ hT) {
    const int c = threadIdx.x, n = blockIdx.x, b = blockIdx.y;
    const int g = c >> 3;
    const float mean = stat[(b * G_ + g) * 2 + 0];
    const float rstd = stat[(b * G_ + g) * 2 + 1];
    float v = (x[((size_t)b * C_ + c) * HW + n] - mean) * rstd * gnw[c] + gnb[c];
    hT[((size_t)b * HW + n) * C_ + c] = (half_t)v;
}

// ---- 3) weight f32 -> f16 ----
__global__ void wcvt_kernel(const float* __restrict__ src, half_t* __restrict__ dst, int n) {
    int i = blockIdx.x * 256 + threadIdx.x;
    if (i < n) dst[i] = (half_t)src[i];
}

// ---- 4/7) projection GEMM, async double-buffered. 256 thr, 128x128 tile ----
// MODE 0: store transposed f16 OutT[b][n][o]  MODE 1: natural f16 Out[b][o][n]
// MODE 2: f32 residual out = x + acc + bias
template <int MODE>
__global__ __launch_bounds__(256, 1) void gemm256_kernel(
        const half_t* __restrict__ A, const half_t* __restrict__ Bm,
        const float* __restrict__ bias, void* __restrict__ outp,
        const float* __restrict__ xres) {
    __shared__ half_t As[2][128][40];
    __shared__ half_t Bs[2][128][40];
    const int t = threadIdx.x;
    const int lane = t & 31, wid = t >> 5;
    const int l = lane & 15, hi = lane >> 4;
    const int wm = wid >> 2, wn = wid & 3;  // 2 x 4 waves; wave tile 64M x 32N
    const int m0g = blockIdx.y * 128;
    const int n0g = blockIdx.x * 128;
    const int b = blockIdx.z;

    unsigned int as_b[2] = { lds_addr32(&As[0][0][0]), lds_addr32(&As[1][0][0]) };
    unsigned int bs_b[2] = { lds_addr32(&Bs[0][0][0]), lds_addr32(&Bs[1][0][0]) };

    auto stage = [&](int buf, int k0) {
#pragma unroll
        for (int u = 0; u < 2; ++u) {
            int chunk = t + 256 * u;
            int row = chunk >> 2, cc = chunk & 3;            // 128 rows x 4 x 16B
            async_cp16(as_b[buf] + row * 80 + cc * 16,
                       A + (size_t)(m0g + row) * C_ + k0 + cc * 8);
            async_cp16(bs_b[buf] + row * 80 + cc * 16,
                       Bm + ((size_t)b * HW + n0g + row) * C_ + k0 + cc * 8);
        }
    };

    v8f acc[4][2];
#pragma unroll
    for (int mt = 0; mt < 4; ++mt)
#pragma unroll
        for (int nt = 0; nt < 2; ++nt) acc[mt][nt] = v8f{};

    stage(0, 0);
#pragma unroll 1
    for (int ks = 0; ks < 8; ++ks) {
        const int cur = ks & 1;
        wait_async0();
        __syncthreads();
        if (ks < 7) stage(cur ^ 1, (ks + 1) * 32);
        // batch all fragment loads, then the WMMA burst (partial dscnt waits overlap)
        v16h bf[2], af[4];
#pragma unroll
        for (int nt = 0; nt < 2; ++nt) bf[nt] = frag_b(&Bs[cur][0][0], 40, wn * 32 + nt * 16, 0, lane);
#pragma unroll
        for (int mt = 0; mt < 4; ++mt) af[mt] = frag_a(&As[cur][0][0], 40, wm * 64 + mt * 16, 0, lane);
#pragma unroll
        for (int mt = 0; mt < 4; ++mt)
#pragma unroll
            for (int nt = 0; nt < 2; ++nt) acc[mt][nt] = WMMA16(af[mt], bf[nt], acc[mt][nt]);
    }

#pragma unroll
    for (int mt = 0; mt < 4; ++mt) {
        const int mb = m0g + wm * 64 + mt * 16 + hi * 8;
#pragma unroll
        for (int nt = 0; nt < 2; ++nt) {
            const int n = n0g + wn * 32 + nt * 16 + l;
            if (MODE == 0) {
                v8h ph;
#pragma unroll
                for (int r = 0; r < 8; ++r) ph[r] = (half_t)(acc[mt][nt][r] + bias[mb + r]);
                *(v8h*)((half_t*)outp + ((size_t)b * HW + n) * C_ + mb) = ph;
            } else if (MODE == 1) {
#pragma unroll
                for (int r = 0; r < 8; ++r)
                    ((half_t*)outp)[((size_t)b * C_ + mb + r) * HW + n] =
                        (half_t)(acc[mt][nt][r] + bias[mb + r]);
            } else {
#pragma unroll
                for (int r = 0; r < 8; ++r) {
                    size_t idx = ((size_t)b * C_ + mb + r) * HW + n;
                    ((float*)outp)[idx] = xres[idx] + acc[mt][nt][r] + bias[mb + r];
                }
            }
        }
    }
}

// ---- 5) pass A: per-row softmax stats over j, k-tiles async double-buffered in LDS ----
__global__ __launch_bounds__(256, 1) void row_stats_kernel(
        const half_t* __restrict__ qT, const half_t* __restrict__ kT,
        float* __restrict__ mrow, float* __restrict__ zrrow) {
    __shared__ half_t Kb[2][64][264];  // 64 j-rows x 256 K, pitch 264 halves
    const int t = threadIdx.x, lane = t & 31, wid = t >> 5;
    const int l = lane & 15, hi = lane >> 4;
    const int b = blockIdx.y;
    const int ib = blockIdx.x * 128 + wid * 16;  // this wave's 16 i-rows
    const half_t* qbase = qT + ((size_t)b * HW + ib) * C_;

    unsigned int kb_b[2] = { lds_addr32(&Kb[0][0][0]), lds_addr32(&Kb[1][0][0]) };
    auto stage = [&](int buf, int j0) {
#pragma unroll
        for (int u = 0; u < 8; ++u) {
            int chunk = t + 256 * u;                       // 64 rows x 32 x 16B
            int row = chunk >> 5, cc = chunk & 31;
            async_cp16(kb_b[buf] + row * 528 + cc * 16,
                       kT + ((size_t)b * HW + j0 + row) * C_ + cc * 8);
        }
    };

    v16h afr[8];  // full K=256 of A resident in regs
#pragma unroll
    for (int kk = 0; kk < 8; ++kk) afr[kk] = frag_a(qbase, C_, 0, kk * 32, lane);

    float m[8], z[8];
#pragma unroll
    for (int r = 0; r < 8; ++r) { m[r] = -3.0e38f; z[r] = 0.f; }

    stage(0, 0);
#pragma unroll 1
    for (int jb = 0; jb < HW / 64; ++jb) {
        const int cur = jb & 1;
        wait_async0();
        __syncthreads();
        if (jb + 1 < HW / 64) stage(cur ^ 1, (jb + 1) * 64);
        // software-pipeline the 4 j-tiles of this block
        v16h bf[8];
#pragma unroll
        for (int kk = 0; kk < 8; ++kk) bf[kk] = frag_b(&Kb[cur][0][0], 264, 0, kk * 32, lane);
#pragma unroll
        for (int jt = 0; jt < 4; ++jt) {
            v16h bn[8];
            if (jt < 3) {
#pragma unroll
                for (int kk = 0; kk < 8; ++kk)
                    bn[kk] = frag_b(&Kb[cur][0][0], 264, (jt + 1) * 16, kk * 32, lane);
            }
            v8f s{};
#pragma unroll
            for (int kk = 0; kk < 8; ++kk) s = WMMA16(afr[kk], bf[kk], s);
#pragma unroll
            for (int r = 0; r < 8; ++r) {
                float v = s[r] * SCALE;
                float mn = fmaxf(m[r], v);
                z[r] = z[r] * __expf(m[r] - mn) + __expf(v - mn);
                m[r] = mn;
            }
#pragma unroll
            for (int kk = 0; kk < 8; ++kk) bf[kk] = bn[kk];
        }
    }
    // reduce over the 16 j-lanes of each half-wave
#pragma unroll
    for (int off = 1; off < 16; off <<= 1) {
#pragma unroll
        for (int r = 0; r < 8; ++r) {
            float mo = __shfl_xor(m[r], off, 32);
            float zo = __shfl_xor(z[r], off, 32);
            float mn = fmaxf(m[r], mo);
            z[r] = z[r] * __expf(m[r] - mn) + zo * __expf(mo - mn);
            m[r] = mn;
        }
    }
    if (l == 0) {
#pragma unroll
        for (int r = 0; r < 8; ++r) {
            int i = ib + hi * 8 + r;  // D layout: M = r + hi*8
            mrow[b * HW + i]  = m[r];
            zrrow[b * HW + i] = 1.0f / z[r];
        }
    }
}

// ---- 6) pass B: fused score + normalize + V@P. K resident in LDS; Q,V async double-buffered ----
// WG: 256 thr = 8 waves; output oT[b][j0..j0+64][0..255].
// Score role: wave -> one 16x16 s-tile (it = wid&1, jt = wid>>1).
// GEMM role:  wave tile 64(c) x 32(j): wm = wid>>1, wnn = wid&1.
__global__ __launch_bounds__(256, 1) void attn_out_kernel(
        const half_t* __restrict__ qT, const half_t* __restrict__ kT,
        const half_t* __restrict__ vN, const float* __restrict__ mrow,
        const float* __restrict__ zrrow, half_t* __restrict__ oT) {
    __shared__ half_t Ks[64][264];      // resident: this WG's 64 k-rows, full K=256
    __shared__ half_t Qs[2][32][264];   // double-buffered 32 q-rows
    __shared__ half_t Vs[2][256][40];   // double-buffered v[c][i0..i0+32]
    __shared__ half_t P[64][40];        // P[j_local][i_local]
    const int t = threadIdx.x, lane = t & 31, wid = t >> 5;
    const int l = lane & 15, hi = lane >> 4;
    const int b = blockIdx.y;
    const int j0 = blockIdx.x * 64;
    const int it = wid & 1, jt = wid >> 1;
    const int wm = wid >> 1, wnn = wid & 1;

    unsigned int ks_b = lds_addr32(&Ks[0][0]);
    unsigned int qs_b[2] = { lds_addr32(&Qs[0][0][0]), lds_addr32(&Qs[1][0][0]) };
    unsigned int vs_b[2] = { lds_addr32(&Vs[0][0][0]), lds_addr32(&Vs[1][0][0]) };

    auto stage_qv = [&](int buf, int i0n) {
#pragma unroll
        for (int u = 0; u < 4; ++u) {                       // Q: 32 rows x 32 x 16B
            int chunk = t + 256 * u;
            int row = chunk >> 5, cc = chunk & 31;
            async_cp16(qs_b[buf] + row * 528 + cc * 16,
                       qT + ((size_t)b * HW + i0n + row) * C_ + cc * 8);
        }
#pragma unroll
        for (int u = 0; u < 4; ++u) {                       // V: 256 rows x 4 x 16B
            int chunk = t + 256 * u;
            int row = chunk >> 2, cc = chunk & 3;
            async_cp16(vs_b[buf] + row * 80 + cc * 16,
                       vN + ((size_t)b * C_ + row) * HW + i0n + cc * 8);
        }
    };

    // stage resident K + first Q/V
#pragma unroll
    for (int u = 0; u < 8; ++u) {
        int chunk = t + 256 * u;
        int row = chunk >> 5, cc = chunk & 31;
        async_cp16(ks_b + row * 528 + cc * 16,
                   kT + ((size_t)b * HW + j0 + row) * C_ + cc * 8);
    }
    stage_qv(0, 0);

    v8f acc[4][2];
#pragma unroll
    for (int mt = 0; mt < 4; ++mt)
#pragma unroll
        for (int nt = 0; nt < 2; ++nt) acc[mt][nt] = v8f{};

#pragma unroll 1
    for (int i0 = 0; i0 < HW; i0 += 32) {
        const int cur = (i0 >> 5) & 1;
        wait_async0();
        __syncthreads();                          // staged buffers visible; prev gemm + P reads done
        if (i0 + 32 < HW) stage_qv(cur ^ 1, i0 + 32);

        // score tile s[i, j], K = C, from LDS: batch loads, then WMMA burst
        v16h qf[8], kf[8];
#pragma unroll
        for (int kk = 0; kk < 8; ++kk) {
            qf[kk] = frag_a(&Qs[cur][0][0], 264, it * 16, kk * 32, lane);
            kf[kk] = frag_b(&Ks[0][0],      264, jt * 16, kk * 32, lane);
        }
        v8f s{};
#pragma unroll
        for (int kk = 0; kk < 8; ++kk) s = WMMA16(qf[kk], kf[kk], s);

        // p = exp(s*scale - m_i) / z_i
        v8h ph;
        const int ig = i0 + it * 16 + hi * 8;
#pragma unroll
        for (int r = 0; r < 8; ++r) {
            float mm = mrow[b * HW + ig + r];
            float rz = zrrow[b * HW + ig + r];
            ph[r] = (half_t)(__expf(fmaf(s[r], SCALE, -mm)) * rz);
        }
        *(v8h*)&P[jt * 16 + l][it * 16 + hi * 8] = ph;  // transpose to [j][i]
        __syncthreads();                          // P visible

        // o[c, j] += v[c, i-block] @ P
        v16h pf[2], vf[4];
#pragma unroll
        for (int nt = 0; nt < 2; ++nt) pf[nt] = frag_b(&P[0][0], 40, wnn * 32 + nt * 16, 0, lane);
#pragma unroll
        for (int mt = 0; mt < 4; ++mt) vf[mt] = frag_a(&Vs[cur][0][0], 40, wm * 64 + mt * 16, 0, lane);
#pragma unroll
        for (int mt = 0; mt < 4; ++mt)
#pragma unroll
            for (int nt = 0; nt < 2; ++nt) acc[mt][nt] = WMMA16(vf[mt], pf[nt], acc[mt][nt]);
    }
    // store oT[b][j][c]
#pragma unroll
    for (int mt = 0; mt < 4; ++mt) {
        const int c0 = wm * 64 + mt * 16 + hi * 8;
#pragma unroll
        for (int nt = 0; nt < 2; ++nt) {
            const int j = j0 + wnn * 32 + nt * 16 + l;
            v8h ph;
#pragma unroll
            for (int r = 0; r < 8; ++r) ph[r] = (half_t)acc[mt][nt][r];
            *(v8h*)(oT + ((size_t)b * HW + j) * C_ + c0) = ph;
        }
    }
}

extern "C" void kernel_launch(void* const* d_in, const int* in_sizes, int n_in,
                              void* d_out, int out_size, void* d_ws, size_t ws_size,
                              hipStream_t stream) {
    (void)in_sizes; (void)n_in; (void)out_size; (void)ws_size;
    const float* x   = (const float*)d_in[0];
    const float* gnw = (const float*)d_in[1];
    const float* gnb = (const float*)d_in[2];
    const float* wq  = (const float*)d_in[3];
    const float* bq  = (const float*)d_in[4];
    const float* wk  = (const float*)d_in[5];
    const float* bk  = (const float*)d_in[6];
    const float* wv  = (const float*)d_in[7];
    const float* bv  = (const float*)d_in[8];
    const float* wo  = (const float*)d_in[9];
    const float* bo  = (const float*)d_in[10];
    float* out = (float*)d_out;

    char* ws = (char*)d_ws;
    size_t off = 0;
    auto take = [&](size_t bytes) -> void* {
        void* p = (void*)(ws + off);
        off = (off + bytes + 255) & ~(size_t)255;
        return p;
    };
    const size_t act = (size_t)B_ * HW * C_ * sizeof(half_t);  // 16 MB
    half_t* hT  = (half_t*)take(act);
    half_t* qT  = (half_t*)take(act);
    half_t* kT  = (half_t*)take(act);
    half_t* vN  = (half_t*)take(act);
    half_t* oT  = (half_t*)take(act);
    half_t* wqh = (half_t*)take((size_t)C_ * C_ * sizeof(half_t));
    half_t* wkh = (half_t*)take((size_t)C_ * C_ * sizeof(half_t));
    half_t* wvh = (half_t*)take((size_t)C_ * C_ * sizeof(half_t));
    half_t* woh = (half_t*)take((size_t)C_ * C_ * sizeof(half_t));
    float*  stat  = (float*)take((size_t)B_ * G_ * 2 * sizeof(float));
    float*  mrow  = (float*)take((size_t)B_ * HW * sizeof(float));
    float*  zrrow = (float*)take((size_t)B_ * HW * sizeof(float));

    gn_stats_kernel<<<B_ * G_, 256, 0, stream>>>(x, stat);
    gn_apply_kernel<<<dim3(HW, B_), 256, 0, stream>>>(x, gnw, gnb, stat, hT);

    wcvt_kernel<<<(C_ * C_) / 256, 256, 0, stream>>>(wq, wqh, C_ * C_);
    wcvt_kernel<<<(C_ * C_) / 256, 256, 0, stream>>>(wk, wkh, C_ * C_);
    wcvt_kernel<<<(C_ * C_) / 256, 256, 0, stream>>>(wv, wvh, C_ * C_);
    wcvt_kernel<<<(C_ * C_) / 256, 256, 0, stream>>>(wo, woh, C_ * C_);

    dim3 ggrid(HW / 128, C_ / 128, B_);
    gemm256_kernel<0><<<ggrid, 256, 0, stream>>>(wqh, hT, bq, (void*)qT, nullptr);
    gemm256_kernel<0><<<ggrid, 256, 0, stream>>>(wkh, hT, bk, (void*)kT, nullptr);
    gemm256_kernel<1><<<ggrid, 256, 0, stream>>>(wvh, hT, bv, (void*)vN, nullptr);

    row_stats_kernel<<<dim3(HW / 128, B_), 256, 0, stream>>>(qT, kT, mrow, zrrow);
    attn_out_kernel<<<dim3(HW / 64, B_), 256, 0, stream>>>(qT, kT, vN, mrow, zrrow, oT);

    gemm256_kernel<2><<<ggrid, 256, 0, stream>>>(woh, oT, bo, (void*)out, x);
}
